// VRNN_41231686042332
// MI455X (gfx1250) — matrix operand order repaired and checked by reading, not compiled, not run on hardware
//
#include <hip/hip_runtime.h>
#include <hip/hip_bf16.h>
#include <stdint.h>

#define DEVINL __device__ __forceinline__

typedef __attribute__((ext_vector_type(16))) __bf16   v16bf;
typedef __attribute__((ext_vector_type(8)))  float    v8f;
typedef __attribute__((ext_vector_type(8)))  unsigned v8u;

namespace {
constexpr int B = 512, T = 512, P = 32, H = 256, L = 64;
constexpr int BT = 16;                 // batch rows per workgroup (one WMMA M-tile)

// ---- f32 activation LDS (rows padded to keep 16B alignment / spread banks)
constexpr int ST_H   = H + 8;          // 264
constexpr int ST_HID = H + 8;          // 264
constexpr int ST_G   = 1024;
constexpr int ST_X   = 40;             // xhat rows (K=32)
constexpr int ST_Z   = 72;             // K=64 rows

constexpr int OFF_H   = 0;
constexpr int OFF_C   = OFF_H   + BT * ST_H;    //  4224
constexpr int OFF_HID = OFF_C   + BT * H;       //  8320
constexpr int OFF_G   = OFF_HID + BT * ST_HID;  // 12544 (gates; enc/prior outputs alias)
constexpr int OFF_EX  = OFF_G   + BT * ST_G;    // 28928
constexpr int OFF_Z   = OFF_EX  + BT * ST_Z;    // 30080
constexpr int OFF_EZ  = OFF_Z   + BT * ST_Z;    // 31232
constexpr int OFF_E2  = OFF_EZ  + BT * ST_Z;    // 32384
constexpr int OFF_XH  = OFF_E2  + BT * 64;      // 33408
constexpr int SMEM_F  = OFF_XH  + BT * ST_X;    // 34048 floats

// ---- bf16 A-fragment staging (dwords): fragment k at [k*256 + lane*8 + j]
constexpr int AP_H_DW = 8 * 256;       // packed h (K=256), built once per step
constexpr int AP_T_DW = 10 * 256;      // general staging (max K=320)
} // namespace

DEVINL unsigned bfpack(float lo, float hi) {   // RNE f32 -> packed 2x bf16
  unsigned ul = __float_as_uint(lo), uh = __float_as_uint(hi);
  ul = (ul + 0x7FFFu + ((ul >> 16) & 1u)) >> 16;
  uh = (uh + 0x7FFFu + ((uh >> 16) & 1u)) >> 16;
  return (uh << 16) | ul;
}

DEVINL float relu1(float v) {          // single v_med3_f32
  return __builtin_amdgcn_fmed3f(v, 0.0f, __builtin_inff());
}
DEVINL float sigm(float v) { return 1.0f / (1.0f + __expf(-v)); }
DEVINL float ftanh(float v) { float e = __expf(2.0f * v); return 1.0f - 2.0f / (e + 1.0f); }

// ---- Cooperative pack: f32 LDS rows -> bf16 A fragments (ISA 16-bit A layout).
// Column c < widthA comes from sA, else from sB (folds concat into the pack).
template <int KT>
DEVINL void pack_A(unsigned* __restrict__ dst, const float* __restrict__ sA, int strideA,
                   int widthA, const float* __restrict__ sB, int strideB, int tid) {
#pragma unroll
  for (int i = tid; i < KT * 256; i += 256) {
    int k = i >> 8, r = i & 255;
    int lane = r >> 3, j = r & 7;
    int row = lane & 15;
    int koff = (lane & 16) ? 8 : 0;
    int kk = k * 32 + koff + ((j < 4) ? 2 * j : 16 + 2 * (j - 4));
    float lo = (kk < widthA) ? sA[row * strideA + kk] : sB[row * strideB + kk - widthA];
    float hi = (kk + 1 < widthA) ? sA[row * strideA + kk + 1]
                                 : sB[row * strideB + kk + 1 - widthA];
    dst[i] = bfpack(lo, hi);
  }
}

// Pack x_t (16 x 32, row-major f32 in global) directly into one A fragment.
DEVINL void pack_x_global(unsigned* __restrict__ dst, const float* __restrict__ g,
                          int rowStride, int tid) {
  int lane = tid >> 3, j = tid & 7;
  int row = lane & 15;
  int koff = (lane & 16) ? 8 : 0;
  int kk = koff + ((j < 4) ? 2 * j : 16 + 2 * (j - 4));
  float2 v = *(const float2*)(g + (size_t)row * rowStride + kk);
  dst[tid] = bfpack(v.x, v.y);
}

// ---- M=16 GEMM via v_wmma_f32_16x16x32_bf16. A fragments come prepacked from LDS;
// k-tiles [0,KS) from apA, [KS,KT) from apH (split resolved at compile time).
// `wave` is an SGPR (readfirstlane), so the n-loop is scalar control flow and
// WMMA always executes under full EXEC.
template <int KT, int KS, int NT, bool RELU, bool ACC>
DEVINL void gemm16(const unsigned* __restrict__ apA, const unsigned* __restrict__ apH,
                   float* sOut, int outStride,
                   const unsigned* __restrict__ frags, const float* __restrict__ bias,
                   int wave, int lane) {
  const int ncol  = lane & 15;
  const int mbase = (lane >> 4) << 3;   // C/D: lanes 16-31 hold M+8
  for (int n = wave; n < NT; n += 8) {
    v8f acc;
    const float bv = bias[n * 16 + ncol];
#pragma unroll
    for (int r = 0; r < 8; ++r) acc[r] = bv;
    const unsigned* fb = frags + (size_t)n * KT * 256;
#pragma unroll
    for (int k = 0; k < KT; ++k) {
      if (k + 1 < KT) __builtin_prefetch(fb + (k + 1) * 256 + lane * 8, 0, 3);
      const unsigned* ap = (k < KS) ? (apA + k * 256 + lane * 8)
                                    : (apH + (k - KS) * 256 + lane * 8);
      uint4 a0 = ((const uint4*)ap)[0];
      uint4 a1 = ((const uint4*)ap)[1];
      const uint4* bp = (const uint4*)(fb + k * 256 + lane * 8);
      uint4 b0 = bp[0], b1 = bp[1];
      v8u au; au[0]=a0.x; au[1]=a0.y; au[2]=a0.z; au[3]=a0.w;
              au[4]=a1.x; au[5]=a1.y; au[6]=a1.z; au[7]=a1.w;
      v8u bu; bu[0]=b0.x; bu[1]=b0.y; bu[2]=b0.z; bu[3]=b0.w;
              bu[4]=b1.x; bu[5]=b1.y; bu[6]=b1.z; bu[7]=b1.w;
      v16bf av = __builtin_bit_cast(v16bf, au);
      v16bf bw = __builtin_bit_cast(v16bf, bu);
      acc = __builtin_amdgcn_wmma_f32_16x16x32_bf16(false, av, false, bw,
                                                    (short)0, acc, false, false);
    }
#pragma unroll
    for (int r = 0; r < 8; ++r) {
      float v = acc[r];
      if (RELU) v = relu1(v);
      float* o = sOut + (mbase + r) * outStride + n * 16 + ncol;
      if (ACC) *o += v; else *o = v;
    }
  }
}

// ---- Prep: repack W(out,in) row-major f32 into bf16 fragments in the ISA 16-bit
// B-matrix lane layout. One block = one 32x16 fragment (256 dwords).
__global__ void pack_weights(const float* __restrict__ W, unsigned* __restrict__ dst,
                             int Kin, int Kt) {
  int f = blockIdx.x;                 // f = n_tile*Kt + k_tile
  int n = f / Kt, k = f % Kt;
  int lane = threadIdx.x >> 3;
  int j = threadIdx.x & 7;
  int kk  = k * 32 + ((lane & 16) ? 8 : 0) + ((j < 4) ? 2 * j : 16 + 2 * (j - 4));
  int col = n * 16 + (lane & 15);
  float lo = W[col * Kin + kk];
  float hi = W[col * Kin + kk + 1];
  dst[f * 256 + lane * 8 + j] = bfpack(lo, hi);
}

struct Params {
  const float *x, *eps;
  const float *b_embx1, *b_embx2, *b_enc1, *b_enc2, *b_pr1, *b_pr2;
  const float *b_embz1, *b_embz2, *b_dec1, *b_dec2, *b_ih, *b_hh;
  const unsigned *f_embx1, *f_embx2, *f_enc1, *f_enc2, *f_pr1, *f_pr2;
  const unsigned *f_embz1, *f_embz2, *f_dec1, *f_dec2, *f_ih, *f_hh;
  float *out_rec, *out_zp, *out_zpp;
};

__global__ __launch_bounds__(256) void vrnn_kernel(Params p) {
  __shared__ __align__(16) float    smf[SMEM_F];
  __shared__ __align__(16) unsigned ap_h[AP_H_DW];
  __shared__ __align__(16) unsigned ap_t[AP_T_DW];
  float* s_h   = smf + OFF_H;
  float* s_c   = smf + OFF_C;
  float* s_hid = smf + OFF_HID;
  float* s_g   = smf + OFF_G;
  float* s_enc = smf + OFF_G;            // alias: enc/prior outputs precede gates
  float* s_pr  = smf + OFF_G + BT * 128;
  float* s_ex  = smf + OFF_EX;
  float* s_z   = smf + OFF_Z;
  float* s_ez  = smf + OFF_EZ;
  float* s_e2  = smf + OFF_E2;
  float* s_xh  = smf + OFF_XH;

  const int tid  = threadIdx.x;
  const int lane = tid & 31;
  const int wave = __builtin_amdgcn_readfirstlane(tid >> 5);  // SGPR: scalar n-loops
  const int b0   = blockIdx.x * BT;

  for (int i = tid; i < BT * ST_H; i += 256) s_h[i] = 0.0f;
  for (int i = tid; i < BT * H;    i += 256) s_c[i] = 0.0f;
  __syncthreads();

  for (int t = 0; t < T; ++t) {
    // pack h (reused by enc/dec concat tails, prior, W_hh) and x_t
    pack_A<8>(ap_h, s_h, ST_H, H, s_h, ST_H, tid);
    pack_x_global(ap_t, p.x + ((size_t)b0 * T + t) * P, T * P, tid);
    __syncthreads();
    // ex = relu(x W1^T + b) W2^T + b
    gemm16<1, 1, 16, true,  false>(ap_t, ap_t, s_hid, ST_HID, p.f_embx1, p.b_embx1, wave, lane);
    __syncthreads();
    pack_A<8>(ap_t, s_hid, ST_HID, H, s_hid, ST_HID, tid);
    __syncthreads();
    gemm16<8, 8, 4, false, false>(ap_t, ap_t, s_ex, ST_Z, p.f_embx2, p.b_embx2, wave, lane);
    __syncthreads();
    // enc([ex, h]): pack only the 2 ex tiles; h tiles come from ap_h
    pack_A<2>(ap_t, s_ex, ST_Z, L, s_ex, ST_Z, tid);
    __syncthreads();
    gemm16<10, 2, 16, true,  false>(ap_t, ap_h, s_hid, ST_HID, p.f_enc1, p.b_enc1, wave, lane);
    __syncthreads();
    pack_A<8>(ap_t, s_hid, ST_HID, H, s_hid, ST_HID, tid);
    __syncthreads();
    gemm16<8, 8, 8, false, false>(ap_t, ap_t, s_enc, 128, p.f_enc2, p.b_enc2, wave, lane);
    __syncthreads();
    // z = mu + min(exp(0.5 lv), 100) * eps ; emit z_params
    for (int i = tid; i < BT * L; i += 256) {
      int r = i >> 6, j = i & 63;
      float mu = s_enc[r * 128 + j], lv = s_enc[r * 128 + L + j];
      float sd = fminf(__expf(0.5f * lv), 100.0f);
      float ep = p.eps[((size_t)(b0 + r) * T + t) * L + j];
      s_z[r * ST_Z + j] = mu + sd * ep;
      size_t ob = (((size_t)(b0 + r) * T + t) * 2) * L + j;
      p.out_zp[ob] = mu;
      p.out_zp[ob + L] = lv;
    }
    __syncthreads();
    // prior = mlp(h) (A comes straight from ap_h)
    gemm16<8, 8, 16, true,  false>(ap_h, ap_h, s_hid, ST_HID, p.f_pr1, p.b_pr1, wave, lane);
    __syncthreads();
    pack_A<8>(ap_t, s_hid, ST_HID, H, s_hid, ST_HID, tid);
    __syncthreads();
    gemm16<8, 8, 8, false, false>(ap_t, ap_t, s_pr, 128, p.f_pr2, p.b_pr2, wave, lane);
    __syncthreads();
    for (int i = tid; i < BT * L; i += 256) {
      int r = i >> 6, j = i & 63;
      size_t ob = (((size_t)(b0 + r) * T + t) * 2) * L + j;
      p.out_zpp[ob] = s_pr[r * 128 + j];
      p.out_zpp[ob + L] = s_pr[r * 128 + L + j];
    }
    // ez = mlp(z)
    pack_A<2>(ap_t, s_z, ST_Z, L, s_z, ST_Z, tid);
    __syncthreads();
    gemm16<2, 2, 16, true,  false>(ap_t, ap_t, s_hid, ST_HID, p.f_embz1, p.b_embz1, wave, lane);
    __syncthreads();
    pack_A<8>(ap_t, s_hid, ST_HID, H, s_hid, ST_HID, tid);
    __syncthreads();
    gemm16<8, 8, 4, false, false>(ap_t, ap_t, s_ez, ST_Z, p.f_embz2, p.b_embz2, wave, lane);
    __syncthreads();
    // dec([ez, h])
    pack_A<2>(ap_t, s_ez, ST_Z, L, s_ez, ST_Z, tid);
    __syncthreads();
    gemm16<10, 2, 16, true,  false>(ap_t, ap_h, s_hid, ST_HID, p.f_dec1, p.b_dec1, wave, lane);
    __syncthreads();
    pack_A<8>(ap_t, s_hid, ST_HID, H, s_hid, ST_HID, tid);
    __syncthreads();
    gemm16<8, 8, 2, false, false>(ap_t, ap_t, s_xh, ST_X, p.f_dec2, p.b_dec2, wave, lane);
    __syncthreads();
    // emit x_hat; embx(x_hat)
    for (int i = tid; i < BT * P; i += 256) {
      int r = i >> 5, c = i & 31;
      p.out_rec[((size_t)(b0 + r) * T + t) * P + c] = s_xh[r * ST_X + c];
    }
    pack_A<1>(ap_t, s_xh, ST_X, P, s_xh, ST_X, tid);
    __syncthreads();
    gemm16<1, 1, 16, true,  false>(ap_t, ap_t, s_hid, ST_HID, p.f_embx1, p.b_embx1, wave, lane);
    __syncthreads();
    pack_A<8>(ap_t, s_hid, ST_HID, H, s_hid, ST_HID, tid);
    __syncthreads();
    gemm16<8, 8, 4, false, false>(ap_t, ap_t, s_e2, 64, p.f_embx2, p.b_embx2, wave, lane);
    __syncthreads();
    // lstm_in = [embx(x_hat), ez] -> 4 k-tiles
    pack_A<4>(ap_t, s_e2, 64, L, s_ez, ST_Z, tid);
    __syncthreads();
    // gates = lstm_in W_ih^T + b_ih  (+)=  h W_hh^T + b_hh (same-wave tile ownership)
    gemm16<4, 4, 64, false, false>(ap_t, ap_t, s_g, ST_G, p.f_ih, p.b_ih, wave, lane);
    gemm16<8, 8, 64, false, true >(ap_h, ap_h, s_g, ST_G, p.f_hh, p.b_hh, wave, lane);
    __syncthreads();
    // LSTM cell update
    for (int i = tid; i < BT * H; i += 256) {
      int r = i >> 8, n = i & 255;
      float ig = s_g[r * ST_G + n],         fg = s_g[r * ST_G + H + n];
      float gg = s_g[r * ST_G + 2 * H + n], og = s_g[r * ST_G + 3 * H + n];
      float cn = sigm(fg) * s_c[r * H + n] + sigm(ig) * ftanh(gg);
      float hn = sigm(og) * ftanh(cn);
      s_c[r * H + n] = cn;
      s_h[r * ST_H + n] = hn;
    }
    __syncthreads();
  }
}

extern "C" void kernel_launch(void* const* d_in, const int* in_sizes, int n_in,
                              void* d_out, int out_size, void* d_ws, size_t ws_size,
                              hipStream_t stream) {
  (void)in_sizes; (void)n_in; (void)out_size; (void)ws_size;
  unsigned* ws = (unsigned*)d_ws;

  // {input idx, ws dword offset, Kin, Kt, Nt}
  struct WT { int idx; unsigned off; int Kin, Kt, Nt; };
  const WT tbl[12] = {
    { 2,      0,  32,  1, 16}, { 4,   4096, 256,  8,  4},
    { 6,  12288, 320, 10, 16}, { 8,  53248, 256,  8,  8},
    {10,  69632, 256,  8, 16}, {12, 102400, 256,  8,  8},
    {14, 118784,  64,  2, 16}, {16, 126976, 256,  8,  4},
    {18, 135168, 320, 10, 16}, {20, 176128, 256,  8,  2},
    {22, 180224, 128,  4, 64}, {24, 245760, 256,  8, 64},
  };
  for (int i = 0; i < 12; ++i) {
    dim3 grid(tbl[i].Kt * tbl[i].Nt);
    pack_weights<<<grid, 256, 0, stream>>>((const float*)d_in[tbl[i].idx],
                                           ws + tbl[i].off, tbl[i].Kin, tbl[i].Kt);
  }

  float* out = (float*)d_out;
  Params p;
  p.x = (const float*)d_in[0];   p.eps = (const float*)d_in[1];
  p.b_embx1 = (const float*)d_in[3];  p.b_embx2 = (const float*)d_in[5];
  p.b_enc1  = (const float*)d_in[7];  p.b_enc2  = (const float*)d_in[9];
  p.b_pr1   = (const float*)d_in[11]; p.b_pr2   = (const float*)d_in[13];
  p.b_embz1 = (const float*)d_in[15]; p.b_embz2 = (const float*)d_in[17];
  p.b_dec1  = (const float*)d_in[19]; p.b_dec2  = (const float*)d_in[21];
  p.b_ih    = (const float*)d_in[23]; p.b_hh    = (const float*)d_in[25];
  p.f_embx1 = ws + tbl[0].off;  p.f_embx2 = ws + tbl[1].off;
  p.f_enc1  = ws + tbl[2].off;  p.f_enc2  = ws + tbl[3].off;
  p.f_pr1   = ws + tbl[4].off;  p.f_pr2   = ws + tbl[5].off;
  p.f_embz1 = ws + tbl[6].off;  p.f_embz2 = ws + tbl[7].off;
  p.f_ih    = ws + tbl[10].off; p.f_hh    = ws + tbl[11].off;
  p.f_dec1  = ws + tbl[8].off;  p.f_dec2  = ws + tbl[9].off;
  p.out_rec = out;
  p.out_zp  = out + (size_t)B * T * P;
  p.out_zpp = out + (size_t)B * T * P + (size_t)B * T * 2 * L;

  vrnn_kernel<<<dim3(B / BT), 256, 0, stream>>>(p);
}
